// MultiHeadAttention_55224689492788
// MI455X (gfx1250) — matrix operand order repaired and checked by reading, not compiled, not run on hardware
//
#include <hip/hip_runtime.h>

// ---------------------------------------------------------------------------
// MI455X / gfx1250 multi-head attention forward.
// bf16 WMMA + f32 accumulate; double-buffered Tensor Data Mover (TDM)
// pipelines for all tile movement (DMA overlapped with WMMA).
// ---------------------------------------------------------------------------

typedef __attribute__((ext_vector_type(16))) __bf16       v16bf;
typedef __attribute__((ext_vector_type(8)))  float        v8f;
typedef __attribute__((ext_vector_type(4)))  unsigned int v4u;
typedef __attribute__((ext_vector_type(8)))  unsigned int v8u;

union Frag16 { v16bf v; uint4 q[2]; };
union FragS  { v16bf v; __bf16 e[16]; };

#define D_MODEL 1024
#define T_SEQ   2048
#define NHEADS  16
#define DK      64

// Generic pointer (to a __shared__ object) -> 32-bit LDS byte offset.
typedef __attribute__((address_space(3))) const char lds_char_t;
static __device__ __forceinline__ unsigned lds_byte_off(const void* p) {
  return (unsigned)(unsigned long long)(lds_char_t*)p;
}

// ---------------------------------------------------------------------------
// TDM 2D tile load: global (row-major, stride_elems between rows, bf16 data)
// -> contiguous LDS tile [rows][row_elems]. Wave-level op, TENSORcnt-tracked,
// retires in order -> s_wait_tensorcnt N pipelines against newer issues.
// D# packing per CDNA5 ISA.
// ---------------------------------------------------------------------------
static __device__ __forceinline__ void tdm_load_2d(unsigned lds_off,
                                                   const void* gsrc,
                                                   unsigned row_elems,
                                                   unsigned rows,
                                                   unsigned stride_elems) {
  unsigned long long ga = (unsigned long long)gsrc;
  v4u g0;
  g0.x = 1u;                                    // count=1, user descriptor
  g0.y = lds_off;                               // lds_addr (bytes)
  g0.z = (unsigned)(ga & 0xffffffffu);          // global_addr[31:0]
  g0.w = (unsigned)((ga >> 32) & 0x01ffffffu)   // global_addr[56:32]
         | (2u << 30);                          // type=2 ("image")
  v8u g1;
  g1.s0 = (1u << 16);                           // data_size=1 (2 bytes)
  g1.s1 = (row_elems & 0xffffu) << 16;          // tensor_dim0[15:0]
  g1.s2 = ((row_elems >> 16) & 0xffffu)         // tensor_dim0[31:16]
        | ((rows & 0xffffu) << 16);             // tensor_dim1[15:0]
  g1.s3 = ((rows >> 16) & 0xffffu)              // tensor_dim1[31:16]
        | ((row_elems & 0xffffu) << 16);        // tile_dim0
  g1.s4 = (rows & 0xffffu);                     // tile_dim1 (tile_dim2=0 -> 2D)
  g1.s5 = stride_elems;                         // tensor_dim0_stride[31:0]
  g1.s6 = 0u;
  g1.s7 = 0u;
  asm volatile("tensor_load_to_lds %0, %1" :: "s"(g0), "s"(g1) : "memory");
}

// ---------------------------------------------------------------------------
// fp32 -> bf16 conversion (one float4 in, 8 bytes out per thread)
// ---------------------------------------------------------------------------
__global__ __launch_bounds__(256)
void f32_to_bf16_kernel(const float* __restrict__ in, __bf16* __restrict__ out, int n4) {
  int i = blockIdx.x * blockDim.x + threadIdx.x;
  if (i < n4) {
    float4 v = ((const float4*)in)[i];
    union { __bf16 h[4]; uint2 u; } pk;
    pk.h[0] = (__bf16)v.x; pk.h[1] = (__bf16)v.y;
    pk.h[2] = (__bf16)v.z; pk.h[3] = (__bf16)v.w;
    *(uint2*)(out + i * 4) = pk.u;
  }
}

// ---------------------------------------------------------------------------
// GEMM: Out[m,e] = sum_k A[m,k] * W[e,k] + bias[e]   (torch Linear: x @ W^T + b)
// Block 256 thr = 8 waves (4 along M x 2 along N), block tile 128x64.
// K-tile 64, double-buffered TDM: tile i+1 streams in while tile i computes.
// Each wave: 2x2 WMMA tiles x 2 K-substeps = 8 v_wmma per K-tile.
// ---------------------------------------------------------------------------
template<bool OUT_F32>
__global__ __launch_bounds__(256)
void gemm_xwT_kernel(const __bf16* __restrict__ A,
                     const __bf16* __restrict__ W,
                     const float*  __restrict__ bias,
                     void* __restrict__ Out)
{
  __shared__ __bf16 lA[2][128][64];   // 32 KB
  __shared__ __bf16 lB[2][64][64];    // 16 KB

  const int tid  = threadIdx.x;
  const int lane = tid & 31;
  const int wave = tid >> 5;
  const int l16  = lane & 15;
  const int half = lane >> 4;
  const int wm   = wave & 3;
  const int wn   = wave >> 2;
  const int m0   = blockIdx.x * 128;
  const int n0   = blockIdx.y * 64;

  const unsigned lA_off0 = lds_byte_off(&lA[0][0][0]);
  const unsigned lA_off1 = lds_byte_off(&lA[1][0][0]);
  const unsigned lB_off0 = lds_byte_off(&lB[0][0][0]);
  const unsigned lB_off1 = lds_byte_off(&lB[1][0][0]);

  v8f acc[2][2];
#pragma unroll
  for (int fn = 0; fn < 2; ++fn) {
    float bv = bias[n0 + wn * 32 + fn * 16 + l16];
#pragma unroll
    for (int fm = 0; fm < 2; ++fm)
#pragma unroll
      for (int j = 0; j < 8; ++j) acc[fm][fn][j] = bv;
  }

  // Prologue: start tile 0 DMA.
  if (wave == 0) {
    tdm_load_2d(lA_off0, A + (size_t)m0 * D_MODEL, 64, 128, D_MODEL);
    tdm_load_2d(lB_off0, W + (size_t)n0 * D_MODEL, 64,  64, D_MODEL);
  }

  for (int k0 = 0; k0 < D_MODEL; k0 += 64) {
    const int cur = (k0 >> 6) & 1;
    __syncthreads();   // all waves done reading the buffer about to be refilled
    if (wave == 0) {
      if (k0 + 64 < D_MODEL) {
        tdm_load_2d(cur ? lA_off0 : lA_off1,
                    A + (size_t)m0 * D_MODEL + (k0 + 64), 64, 128, D_MODEL);
        tdm_load_2d(cur ? lB_off0 : lB_off1,
                    W + (size_t)n0 * D_MODEL + (k0 + 64), 64,  64, D_MODEL);
        __builtin_amdgcn_s_wait_tensorcnt(2);   // tile k0 landed; next in flight
      } else {
        __builtin_amdgcn_s_wait_tensorcnt(0);
      }
    }
    __syncthreads();

#pragma unroll
    for (int ks = 0; ks < 2; ++ks) {
      // A fragments (16x32: lane<16 -> K 0..7,16..23 ; lane>=16 -> K 8..15,24..31)
      Frag16 af[2];
#pragma unroll
      for (int fm = 0; fm < 2; ++fm) {
        const __bf16* p = &lA[cur][wm * 32 + fm * 16 + l16][ks * 32 + half * 8];
        af[fm].q[0] = *(const uint4*)p;
        af[fm].q[1] = *(const uint4*)(p + 16);
      }
      // B fragments (32x16: col = l16; lane<16 -> K 0..7,16..23 ; lane>=16 -> K 8..15,24..31)
      Frag16 bf[2];
#pragma unroll
      for (int fn = 0; fn < 2; ++fn) {
        const __bf16* p = &lB[cur][wn * 32 + fn * 16 + l16][ks * 32 + half * 8];
        bf[fn].q[0] = *(const uint4*)p;
        bf[fn].q[1] = *(const uint4*)(p + 16);
      }
#pragma unroll
      for (int fm = 0; fm < 2; ++fm)
#pragma unroll
        for (int fn = 0; fn < 2; ++fn)
          acc[fm][fn] = __builtin_amdgcn_wmma_f32_16x16x32_bf16(
              false, af[fm].v, false, bf[fn].v, (short)0, acc[fm][fn], false, false);
    }
  }

#pragma unroll
  for (int fm = 0; fm < 2; ++fm)
#pragma unroll
    for (int fn = 0; fn < 2; ++fn)
#pragma unroll
      for (int j = 0; j < 8; ++j) {
        size_t row = (size_t)(m0 + wm * 32 + fm * 16 + j + 8 * half);
        int    col = n0 + wn * 32 + fn * 16 + l16;
        if (OUT_F32)
          ((float*)Out)[row * D_MODEL + col] = acc[fm][fn][j];
        else
          ((__bf16*)Out)[row * D_MODEL + col] = (__bf16)acc[fm][fn][j];
      }
}

// ---------------------------------------------------------------------------
// Flash-style attention. Grid: (T/128, H, B). Block 256 thr = 8 waves; each
// wave owns 16 queries. K/V tiles (32x64 bf16, row-major) stream via
// double-buffered TDM; S = Q K^T (4 wmma), online softmax (f32), P converted
// to A-layout via per-wave LDS tile, acc += P V (4 wmma, V B-frags gathered
// strided from the row-major tile). Output head-concatenated [B*T,1024] bf16.
// ---------------------------------------------------------------------------
__global__ __launch_bounds__(256)
void attention_kernel(const __bf16* __restrict__ Q,
                      const __bf16* __restrict__ K,
                      const __bf16* __restrict__ V,
                      __bf16* __restrict__ Oc)
{
  __shared__ __bf16 lK[2][32][64];    // 8 KB
  __shared__ __bf16 lV[2][32][64];    // 8 KB
  __shared__ __bf16 lP[8][16][32];    // 8 KB, per-wave P staging

  const int b    = blockIdx.z;
  const int h    = blockIdx.y;
  const int tid  = threadIdx.x;
  const int lane = tid & 31;
  const int wave = tid >> 5;
  const int l16  = lane & 15;
  const int half = lane >> 4;
  const int q0   = blockIdx.x * 128 + wave * 16;
  const float scale = 0.03125f;       // 1/sqrt(d_model) = 1/32

  const unsigned lK_off0 = lds_byte_off(&lK[0][0][0]);
  const unsigned lK_off1 = lds_byte_off(&lK[1][0][0]);
  const unsigned lV_off0 = lds_byte_off(&lV[0][0][0]);
  const unsigned lV_off1 = lds_byte_off(&lV[1][0][0]);

  // Load the wave's Q fragments once (reused across all 64 key steps).
  const __bf16* Qrow = Q + ((size_t)(b * T_SEQ + q0 + l16)) * D_MODEL + h * DK;
  Frag16 aq[2];
#pragma unroll
  for (int ks = 0; ks < 2; ++ks) {
    const __bf16* p = Qrow + ks * 32 + half * 8;
    aq[ks].q[0] = *(const uint4*)p;
    aq[ks].q[1] = *(const uint4*)(p + 16);
  }

  float mrow[8], lrow[8];
  v8f acc[4];
  v8f zero = {};
#pragma unroll
  for (int f = 0; f < 4; ++f) acc[f] = zero;
#pragma unroll
  for (int j = 0; j < 8; ++j) { mrow[j] = -3.0e38f; lrow[j] = 0.0f; }

  const size_t hbase = (size_t)(b * T_SEQ) * D_MODEL + h * DK;

  // Prologue: start tile 0 DMA.
  if (wave == 0) {
    tdm_load_2d(lK_off0, K + hbase, 64, 32, D_MODEL);
    tdm_load_2d(lV_off0, V + hbase, 64, 32, D_MODEL);
  }

  for (int s0 = 0; s0 < T_SEQ; s0 += 32) {
    const int cur = (s0 >> 5) & 1;
    __syncthreads();   // all waves done reading the buffer about to be refilled
    if (wave == 0) {
      if (s0 + 32 < T_SEQ) {
        size_t gn = hbase + (size_t)(s0 + 32) * D_MODEL;
        tdm_load_2d(cur ? lK_off0 : lK_off1, K + gn, 64, 32, D_MODEL);
        tdm_load_2d(cur ? lV_off0 : lV_off1, V + gn, 64, 32, D_MODEL);
        __builtin_amdgcn_s_wait_tensorcnt(2);   // tile s0 landed; next in flight
      } else {
        __builtin_amdgcn_s_wait_tensorcnt(0);
      }
    }
    __syncthreads();

    // S = Q K^T for 32 keys: two 16x16 C tiles, each accumulating over d_k=64.
    v8f s[2];
#pragma unroll
    for (int t2 = 0; t2 < 2; ++t2) {
      v8f sa = zero;
#pragma unroll
      for (int ks = 0; ks < 2; ++ks) {
        Frag16 bk;
        const __bf16* p = &lK[cur][t2 * 16 + l16][ks * 32 + half * 8];
        bk.q[0] = *(const uint4*)p;
        bk.q[1] = *(const uint4*)(p + 16);
        sa = __builtin_amdgcn_wmma_f32_16x16x32_bf16(
            false, aq[ks].v, false, bk.v, (short)0, sa, false, false);
      }
      s[t2] = sa;
    }

    // Online softmax update (row r = j + 8*half lives across lanes {half*16..+15})
    float corr[8];
#pragma unroll
    for (int j = 0; j < 8; ++j) {
      float v0 = s[0][j] * scale;
      float v1 = s[1][j] * scale;
      float tmax = fmaxf(v0, v1);
#pragma unroll
      for (int m = 8; m >= 1; m >>= 1) tmax = fmaxf(tmax, __shfl_xor(tmax, m, 32));
      float nm = fmaxf(mrow[j], tmax);
      float c  = __expf(mrow[j] - nm);
      mrow[j]  = nm;
      float p0 = __expf(v0 - nm);
      float p1 = __expf(v1 - nm);
      s[0][j] = p0; s[1][j] = p1;
      float rs = p0 + p1;
#pragma unroll
      for (int m = 8; m >= 1; m >>= 1) rs += __shfl_xor(rs, m, 32);
      lrow[j] = lrow[j] * c + rs;
      corr[j] = c;
    }
#pragma unroll
    for (int f = 0; f < 4; ++f)
#pragma unroll
      for (int j = 0; j < 8; ++j) acc[f][j] *= corr[j];

    // P: C layout f32 -> A layout bf16 via per-wave LDS tile.
#pragma unroll
    for (int j = 0; j < 8; ++j) {
      lP[wave][j + 8 * half][l16]      = (__bf16)s[0][j];
      lP[wave][j + 8 * half][l16 + 16] = (__bf16)s[1][j];
    }
    __syncthreads();   // uniform; makes the cross-lane LDS traffic visible

    Frag16 pa;
    {
      const __bf16* p = &lP[wave][l16][half * 8];
      pa.q[0] = *(const uint4*)p;
      pa.q[1] = *(const uint4*)(p + 16);
    }

    // acc += P (16x32) x V (32x64): 4 B-fragments gathered from row-major V
    // tile: b[i] = V[s = (i&7) + 8*half + 16*(i>>3)][d = f*16 + l16].
#pragma unroll
    for (int f = 0; f < 4; ++f) {
      FragS bv;
#pragma unroll
      for (int i = 0; i < 16; ++i) bv.e[i] = lV[cur][(i & 7) + 8 * half + (i >> 3) * 16][f * 16 + l16];
      acc[f] = __builtin_amdgcn_wmma_f32_16x16x32_bf16(
          false, pa.v, false, bv.v, (short)0, acc[f], false, false);
    }
  }

  // Normalize and store head-concatenated bf16 output.
#pragma unroll
  for (int j = 0; j < 8; ++j) lrow[j] = 1.0f / lrow[j];
#pragma unroll
  for (int f = 0; f < 4; ++f)
#pragma unroll
    for (int j = 0; j < 8; ++j) {
      size_t row = (size_t)(b * T_SEQ + q0 + j + 8 * half);
      int    col = h * DK + f * 16 + l16;
      Oc[row * D_MODEL + col] = (__bf16)(acc[f][j] * lrow[j]);
    }
}

// ---------------------------------------------------------------------------
// Launcher. Inputs: x, Wq, bq, Wk, bk, Wv, bv, Wo, bo (all fp32).
// Workspace layout (bytes): xb[0,8M) wq[8M) wk[10M) wv[12M) wo[14M)
//                           Q[16M) K[24M) V[32M) attn[40M,48M)
// ---------------------------------------------------------------------------
extern "C" void kernel_launch(void* const* d_in, const int* in_sizes, int n_in,
                              void* d_out, int out_size, void* d_ws, size_t ws_size,
                              hipStream_t stream)
{
  const float* x  = (const float*)d_in[0];
  const float* Wq = (const float*)d_in[1];
  const float* bq = (const float*)d_in[2];
  const float* Wk = (const float*)d_in[3];
  const float* bk = (const float*)d_in[4];
  const float* Wv = (const float*)d_in[5];
  const float* bv = (const float*)d_in[6];
  const float* Wo = (const float*)d_in[7];
  const float* bo = (const float*)d_in[8];

  char* ws = (char*)d_ws;
  const size_t MB = 1024ull * 1024ull;
  __bf16* xb  = (__bf16*)(ws);
  __bf16* wqb = (__bf16*)(ws +  8 * MB);
  __bf16* wkb = (__bf16*)(ws + 10 * MB);
  __bf16* wvb = (__bf16*)(ws + 12 * MB);
  __bf16* wob = (__bf16*)(ws + 14 * MB);
  __bf16* Qb  = (__bf16*)(ws + 16 * MB);
  __bf16* Kb  = (__bf16*)(ws + 24 * MB);
  __bf16* Vb  = (__bf16*)(ws + 32 * MB);
  __bf16* Ob  = (__bf16*)(ws + 40 * MB);

  f32_to_bf16_kernel<<<4096, 256, 0, stream>>>(x,  xb,  (2 * T_SEQ * D_MODEL) / 4);
  f32_to_bf16_kernel<<<1024, 256, 0, stream>>>(Wq, wqb, (D_MODEL * D_MODEL) / 4);
  f32_to_bf16_kernel<<<1024, 256, 0, stream>>>(Wk, wkb, (D_MODEL * D_MODEL) / 4);
  f32_to_bf16_kernel<<<1024, 256, 0, stream>>>(Wv, wvb, (D_MODEL * D_MODEL) / 4);
  f32_to_bf16_kernel<<<1024, 256, 0, stream>>>(Wo, wob, (D_MODEL * D_MODEL) / 4);

  dim3 gg(32, 16);   // M/128 x 1024/64
  gemm_xwT_kernel<false><<<gg, 256, 0, stream>>>(xb, wqb, bq, (void*)Qb);
  gemm_xwT_kernel<false><<<gg, 256, 0, stream>>>(xb, wkb, bk, (void*)Kb);
  gemm_xwT_kernel<false><<<gg, 256, 0, stream>>>(xb, wvb, bv, (void*)Vb);

  dim3 ga(T_SEQ / 128, NHEADS, 2);
  attention_kernel<<<ga, 256, 0, stream>>>(Qb, Kb, Vb, Ob);

  gemm_xwT_kernel<true><<<gg, 256, 0, stream>>>(Ob, wob, bo, d_out);
}
